// GraphTransformerVN_14267881357529
// MI455X (gfx1250) — matrix operand order repaired and checked by reading, not compiled, not run on hardware
//
#include <hip/hip_runtime.h>
#include <math.h>

// ---------------- types for WMMA ----------------
typedef __attribute__((ext_vector_type(16))) __bf16 v16bf;
typedef __attribute__((ext_vector_type(8)))  __bf16 v8bf;
typedef __attribute__((ext_vector_type(8)))  float  v8f;

#define N_NODES 10000
#define N_EDGES 60000
#define IN_DIM  768
#define HID     256
#define HEADS   8
#define CH      256          // per-head channels
#define HC      (HEADS*CH)   // 2048
#define LAYERS  4
#define ETOT    (N_EDGES + N_NODES)
#define EPS_LN  1e-5f
#define SLOPE   0.2f

__device__ __forceinline__ float gelu_f(float v) {
    return 0.5f * v * (1.0f + erff(v * 0.70710678118654752440f));
}

__device__ __forceinline__ void atomicMaxFloat(float* addr, float v) {
    if (v >= 0.0f) atomicMax((int*)addr, __float_as_int(v));
    else           atomicMin((unsigned int*)addr, __float_as_uint(v));
}

// ---------------- bf16 WMMA GEMM ----------------
// out[m,n] = act( sum_k A[m,k]*W[n,k] + bias[n] )
// A: [M,K] bf16 row-major, W: [Ncols,K] bf16 row-major.
// One wave computes a 16x128 output tile: 1 A fragment reused across 8 WMMAs per k-step.
// M%16==0, Ncols%128==0, K%32==0 guaranteed by caller.
__global__ void __launch_bounds__(32)
wmma_gemm_kernel(const __bf16* __restrict__ A, const __bf16* __restrict__ W,
                 const float* __restrict__ bias,
                 float* __restrict__ outF, __bf16* __restrict__ outB,
                 int M, int Ncols, int K, int act)
{
    const int lane = threadIdx.x;        // 0..31
    const int tn   = blockIdx.x;         // 128-column tile
    const int tm   = blockIdx.y;         // 16-row tile
    const int half = lane >> 4;          // 0: lanes 0-15, 1: lanes 16-31
    const int l15  = lane & 15;

    const int m  = tm * 16 + l15;        // A fragment row for this lane
    const int n0 = tn * 128 + l15;       // first B fragment column for this lane

    // A fragment: lanes 0-15 need K = kb+0..15; lanes 16-31 K = kb+16..31
    const __bf16* arow = A + (size_t)m * K + (half ? 16 : 0);
    // B fragment: lanes 0-15 need K = kb+0..15; lanes 16-31 K = kb+16..31 (contiguous 32B)
    const __bf16* brow = W + (size_t)n0 * K + (half ? 16 : 0);
    const size_t  bstride = (size_t)16 * K;   // 16 output columns per subtile

    v8f acc[8] = {};
    for (int kb = 0; kb < K; kb += 32) {
        v16bf a;
        ((v8bf*)&a)[0] = *(const v8bf*)(arow + kb);        // 16B contiguous
        ((v8bf*)&a)[1] = *(const v8bf*)(arow + kb + 8);    // 16B contiguous
#pragma unroll
        for (int j = 0; j < 8; ++j) {
            v16bf b = *(const v16bf*)(brow + (size_t)j * bstride + kb);  // 32B contiguous
            acc[j] = __builtin_amdgcn_wmma_f32_16x16x32_bf16(
                         false, a, false, b, (short)0, acc[j], false, false);
        }
    }

    const int rbase = tm * 16 + (half ? 8 : 0);   // D layout: lanes 16-31 hold rows +8
#pragma unroll
    for (int j = 0; j < 8; ++j) {
        const int n = n0 + j * 16;
        const float bn = bias ? bias[n] : 0.0f;
#pragma unroll
        for (int r = 0; r < 8; ++r) {
            float v = acc[j][r] + bn;
            if (act) v = gelu_f(v);
            size_t o = (size_t)(rbase + r) * Ncols + n;
            if (outF) outF[o] = v;
            if (outB) outB[o] = (__bf16)v;
        }
    }
}

// ---------------- elementwise helpers ----------------
__global__ void f2bf_kernel(const float* __restrict__ src, __bf16* __restrict__ dst, int n) {
    int i = blockIdx.x * blockDim.x + threadIdx.x;
    if (i < n) dst[i] = (__bf16)src[i];
}

__global__ void init_kernel(float* emax, float* den, float* xloc, int nh, int nhid) {
    int i = blockIdx.x * blockDim.x + threadIdx.x;
    if (i < nhid) xloc[i] = 0.0f;
    if (i < nh) { emax[i] = -INFINITY; den[i] = 0.0f; }
}

// ---------------- GATv2 edge kernels ----------------
// Block per edge, 256 threads. Thread c handles channel c for all 8 heads (coalesced reads),
// 8 parallel LDS tree reductions, then 8 lanes emit scores + atomic max into emax[dst,h].
__global__ void __launch_bounds__(CH)
edge_score_kernel(const __bf16* __restrict__ xl, const __bf16* __restrict__ xr,
                  const float* __restrict__ att, const int* __restrict__ ei,
                  float* __restrict__ ebuf, float* __restrict__ emax)
{
    __shared__ float part[HEADS][CH];
    int e = blockIdx.x;
    int s = (e < N_EDGES) ? ei[e]           : (e - N_EDGES);
    int d = (e < N_EDGES) ? ei[N_EDGES + e] : (e - N_EDGES);
    int c = threadIdx.x;
    const __bf16* pl = xl + (size_t)s * HC;
    const __bf16* pr = xr + (size_t)d * HC;
#pragma unroll
    for (int h = 0; h < HEADS; ++h) {
        float m = (float)pl[h * CH + c] + (float)pr[h * CH + c];
        m = (m > 0.0f) ? m : SLOPE * m;
        part[h][c] = m * att[h * CH + c];
    }
    __syncthreads();
    for (int st = CH / 2; st > 0; st >>= 1) {
        if (c < st) {
#pragma unroll
            for (int h = 0; h < HEADS; ++h) part[h][c] += part[h][c + st];
        }
        __syncthreads();
    }
    if (c < HEADS) {
        float v = part[c][0];
        ebuf[e * HEADS + c] = v;
        atomicMaxFloat(&emax[d * HEADS + c], v);
    }
}

// a = exp(e - emax[dst]); store in place; atomic add into denom[dst,h]
__global__ void edge_exp_kernel(float* __restrict__ ebuf, const float* __restrict__ emax,
                                float* __restrict__ den, const int* __restrict__ ei)
{
    int t = blockIdx.x * blockDim.x + threadIdx.x;
    if (t >= ETOT * HEADS) return;
    int e = t >> 3, h = t & 7;
    int d = (e < N_EDGES) ? ei[N_EDGES + e] : (e - N_EDGES);
    float a = expf(ebuf[t] - emax[d * HEADS + h]);
    ebuf[t] = a;
    atomicAdd(&den[d * HEADS + h], a);
}

// block per edge, thread per channel; reduce over heads in registers -> one atomic per (edge,c)
__global__ void __launch_bounds__(CH)
edge_agg_kernel(const __bf16* __restrict__ xl, const float* __restrict__ ebuf,
                const float* __restrict__ den, const int* __restrict__ ei,
                float* __restrict__ xloc)
{
    __shared__ float alpha[HEADS];
    int e = blockIdx.x;
    int s = (e < N_EDGES) ? ei[e]           : (e - N_EDGES);
    int d = (e < N_EDGES) ? ei[N_EDGES + e] : (e - N_EDGES);
    if (threadIdx.x < HEADS)
        alpha[threadIdx.x] = ebuf[e * HEADS + threadIdx.x] / den[d * HEADS + threadIdx.x];
    __syncthreads();
    int c = threadIdx.x;
    const __bf16* pl = xl + (size_t)s * HC + c;
    float acc = 0.0f;
#pragma unroll
    for (int h = 0; h < HEADS; ++h)
        acc += alpha[h] * (float)pl[h * CH];
    atomicAdd(&xloc[(size_t)d * HID + c], acc * (1.0f / HEADS));
}

// ---------------- LayerNorm(x + add [+ eb]) ; block per node, 256 threads ----------------
__global__ void __launch_bounds__(HID)
ln_residual_kernel(const float* __restrict__ xin, const float* __restrict__ add,
                   const float* __restrict__ eb, const float* __restrict__ g,
                   const float* __restrict__ b, float* __restrict__ outF,
                   __bf16* __restrict__ outB)
{
    __shared__ float red[HID];
    int n = blockIdx.x, c = threadIdx.x;
    float v = xin[(size_t)n * HID + c] + add[(size_t)n * HID + c] + (eb ? eb[c] : 0.0f);
    red[c] = v; __syncthreads();
    for (int s = HID / 2; s > 0; s >>= 1) { if (c < s) red[c] += red[c + s]; __syncthreads(); }
    float mean = red[0] * (1.0f / HID);
    __syncthreads();
    float dv = v - mean;
    red[c] = dv * dv; __syncthreads();
    for (int s = HID / 2; s > 0; s >>= 1) { if (c < s) red[c] += red[c + s]; __syncthreads(); }
    float var = red[0] * (1.0f / HID);
    float r = dv * rsqrtf(var + EPS_LN) * g[c] + b[c];
    outF[(size_t)n * HID + c] = r;
    if (outB) outB[(size_t)n * HID + c] = (__bf16)r;
}

// ---------------- virtual node ----------------
__global__ void __launch_bounds__(256)
col_mean_kernel(const float* __restrict__ x, float* __restrict__ vnin)
{
    __shared__ float red[256];
    int c = blockIdx.x;
    float s = 0.0f;
    for (int n = threadIdx.x; n < N_NODES; n += 256) s += x[(size_t)n * HID + c];
    red[threadIdx.x] = s; __syncthreads();
    for (int s2 = 128; s2 > 0; s2 >>= 1) { if (threadIdx.x < s2) red[threadIdx.x] += red[threadIdx.x + s2]; __syncthreads(); }
    if (threadIdx.x == 0) vnin[c] = red[0] * (1.0f / N_NODES);
}

__global__ void __launch_bounds__(512)
vn_mlp_kernel(const float* __restrict__ vnin, float* __restrict__ vn,
              const float* __restrict__ V1, const float* __restrict__ v1b,
              const float* __restrict__ V2, const float* __restrict__ v2b,
              const float* __restrict__ g, const float* __restrict__ b)
{
    __shared__ float vin[HID];
    __shared__ float h1[2 * HID];
    __shared__ float red[512];
    int t = threadIdx.x;
    if (t < HID) vin[t] = vnin[t];
    __syncthreads();
    { // h1 = gelu(vnin @ V1^T + v1b), 512 outputs
        float a = v1b[t];
        const float* w = V1 + (size_t)t * HID;
        for (int c = 0; c < HID; ++c) a += vin[c] * w[c];
        h1[t] = gelu_f(a);
    }
    __syncthreads();
    float u = 0.0f;
    if (t < HID) { // vn_up = h1 @ V2^T + v2b, 256 outputs
        float a = v2b[t];
        const float* w = V2 + (size_t)t * (2 * HID);
        for (int k = 0; k < 2 * HID; ++k) a += h1[k] * w[k];
        u = vn[t] + a;
    }
    red[t] = (t < HID) ? u : 0.0f; __syncthreads();
    for (int s = 256; s > 0; s >>= 1) { if (t < s) red[t] += red[t + s]; __syncthreads(); }
    float mean = red[0] * (1.0f / HID);
    __syncthreads();
    float dv = u - mean;
    red[t] = (t < HID) ? dv * dv : 0.0f; __syncthreads();
    for (int s = 256; s > 0; s >>= 1) { if (t < s) red[t] += red[t + s]; __syncthreads(); }
    float var = red[0] * (1.0f / HID);
    if (t < HID) vn[t] = dv * rsqrtf(var + EPS_LN) * g[t] + b[t];
}

// xc = bf16([x, broadcast(vn)]) as [N, 512]
__global__ void concat_kernel(const float* __restrict__ x, const float* __restrict__ vn,
                              __bf16* __restrict__ xc)
{
    int i = blockIdx.x * blockDim.x + threadIdx.x;
    if (i >= N_NODES * 2 * HID) return;
    int n = i >> 9, c = i & 511;
    xc[i] = (__bf16)((c < HID) ? x[(size_t)n * HID + c] : vn[c - HID]);
}

// logits = h @ O2^T + o2b ; h is f32 [N,256], O2 [2,256]
__global__ void final_logits_kernel(const float* __restrict__ h, const float* __restrict__ O2,
                                    const float* __restrict__ o2b, float* __restrict__ out)
{
    int n = blockIdx.x * blockDim.x + threadIdx.x;
    if (n >= N_NODES) return;
    const float* hp = h + (size_t)n * HID;
    float a0 = o2b[0], a1 = o2b[1];
    for (int c = 0; c < HID; ++c) {
        float v = hp[c];
        a0 += v * O2[c];
        a1 += v * O2[HID + c];
    }
    out[n * 2 + 0] = a0;
    out[n * 2 + 1] = a1;
}

// ---------------- host orchestration ----------------
static inline void run_gemm(const __bf16* A, const __bf16* W, const float* bias,
                            float* outF, __bf16* outB, int M, int Ncols, int K, int act,
                            hipStream_t s)
{
    dim3 grid(Ncols / 128, M / 16);
    wmma_gemm_kernel<<<grid, 32, 0, s>>>(A, W, bias, outF, outB, M, Ncols, K, act);
}

static inline void cvt(const float* src, __bf16* dst, int n, hipStream_t s) {
    f2bf_kernel<<<(n + 255) / 256, 256, 0, s>>>(src, dst, n);
}

extern "C" void kernel_launch(void* const* d_in, const int* in_sizes, int n_in,
                              void* d_out, int out_size, void* d_ws, size_t ws_size,
                              hipStream_t stream)
{
    (void)in_sizes; (void)n_in; (void)out_size; (void)ws_size;
    // inputs in setup_inputs() order
    const float* x_in = (const float*)d_in[0];
    const int*   ei   = (const int*)d_in[1];
    const float* W_in = (const float*)d_in[2];
    const float* b_in = (const float*)d_in[3];
    const float* vnode= (const float*)d_in[4];
    const float* Wl   = (const float*)d_in[5];
    const float* bl   = (const float*)d_in[6];
    const float* Wr   = (const float*)d_in[7];
    const float* br   = (const float*)d_in[8];
    const float* att  = (const float*)d_in[9];
    const float* gbias= (const float*)d_in[10];
    const float* V1   = (const float*)d_in[11];
    const float* v1b  = (const float*)d_in[12];
    const float* V2   = (const float*)d_in[13];
    const float* v2b  = (const float*)d_in[14];
    const float* U1   = (const float*)d_in[15];
    const float* u1b  = (const float*)d_in[16];
    const float* U2   = (const float*)d_in[17];
    const float* u2b  = (const float*)d_in[18];
    const float* g1   = (const float*)d_in[19];
    const float* be1  = (const float*)d_in[20];
    const float* g2   = (const float*)d_in[21];
    const float* be2  = (const float*)d_in[22];
    const float* g3   = (const float*)d_in[23];
    const float* be3  = (const float*)d_in[24];
    const float* O1   = (const float*)d_in[25];
    const float* o1b  = (const float*)d_in[26];
    const float* O2   = (const float*)d_in[27];
    const float* o2b  = (const float*)d_in[28];

    // workspace carve (256B aligned); ~117 MB total
    char* base = (char*)d_ws; size_t off = 0;
    auto take = [&](size_t bytes) -> void* {
        void* r = base + off; off += (bytes + 255) & ~(size_t)255; return r;
    };
    float*  x    = (float*) take((size_t)N_NODES * HID * 4);            // activations f32
    __bf16* xb   = (__bf16*)take((size_t)N_NODES * 2 * HID * 2);        // bf16 x [N,256] / xc [N,512]
    __bf16* xlb  = (__bf16*)take((size_t)N_NODES * HC * 2);             // xl bf16 [N,2048]; also input-bf16 & hidden
    __bf16* xrb  = (__bf16*)take((size_t)N_NODES * HC * 2);             // xr bf16 [N,2048]
    float*  ebuf = (float*) take((size_t)ETOT * HEADS * 4);
    float*  emax = (float*) take((size_t)N_NODES * HEADS * 4);
    float*  den  = (float*) take((size_t)N_NODES * HEADS * 4);
    float*  xloc = (float*) take((size_t)N_NODES * HID * 4);            // x_local / xu / O1-out
    __bf16* wb   = (__bf16*)take((size_t)HC * HID * 2);                 // weight bf16 scratch (max 2048x256)
    float*  vn   = (float*) take(HID * 4);
    float*  vnin = (float*) take(HID * 4);

    hipMemcpyAsync(vn, vnode, HID * 4, hipMemcpyDeviceToDevice, stream);

    // ---- input projection: x = x_in @ W_in^T + b_in ----
    cvt(x_in, xlb, N_NODES * IN_DIM, stream);                 // reuse xl buffer for bf16(x_in)
    cvt(W_in, wb, HID * IN_DIM, stream);
    run_gemm(xlb, wb, b_in, x, xb, N_NODES, HID, IN_DIM, 0, stream);   // also emits xb = bf16(x)

    const int egrid = (ETOT * HEADS + 255) / 256;
    for (int i = 0; i < LAYERS; ++i) {
        // xl = x @ Wl^T + bl ; xr = x @ Wr^T + br   (bf16 outputs for edge kernels)
        cvt(Wl + (size_t)i * HC * HID, wb, HC * HID, stream);
        run_gemm(xb, wb, bl + (size_t)i * HC, nullptr, xlb, N_NODES, HC, HID, 0, stream);
        cvt(Wr + (size_t)i * HC * HID, wb, HC * HID, stream);
        run_gemm(xb, wb, br + (size_t)i * HC, nullptr, xrb, N_NODES, HC, HID, 0, stream);

        // segment softmax attention + aggregation
        init_kernel<<<(N_NODES * HID + 255) / 256, 256, 0, stream>>>(emax, den, xloc,
                                                                     N_NODES * HEADS, N_NODES * HID);
        edge_score_kernel<<<ETOT, CH, 0, stream>>>(xlb, xrb, att + (size_t)i * HC, ei, ebuf, emax);
        edge_exp_kernel<<<egrid, 256, 0, stream>>>(ebuf, emax, den, ei);
        edge_agg_kernel<<<ETOT, CH, 0, stream>>>(xlb, ebuf, den, ei, xloc);

        // x = LN(x + x_local + gbias)
        ln_residual_kernel<<<N_NODES, HID, 0, stream>>>(x, xloc, gbias + (size_t)i * HID,
                                                        g1 + (size_t)i * HID, be1 + (size_t)i * HID,
                                                        x, nullptr);

        // virtual node update
        col_mean_kernel<<<HID, 256, 0, stream>>>(x, vnin);
        vn_mlp_kernel<<<1, 512, 0, stream>>>(vnin, vn,
                                             V1 + (size_t)i * 2 * HID * HID, v1b + (size_t)i * 2 * HID,
                                             V2 + (size_t)i * HID * 2 * HID, v2b + (size_t)i * HID,
                                             g2 + (size_t)i * HID, be2 + (size_t)i * HID);

        // node MLP with broadcast VN: xu = gelu([x,vn] @ U1^T + u1b) @ U2^T + u2b
        concat_kernel<<<(N_NODES * 2 * HID + 255) / 256, 256, 0, stream>>>(x, vn, xb); // xb as [N,512]
        cvt(U1 + (size_t)i * 2 * HID * 2 * HID, wb, 2 * HID * 2 * HID, stream);
        run_gemm(xb, wb, u1b + (size_t)i * 2 * HID, nullptr, xlb /*hidden bf16 [N,512]*/,
                 N_NODES, 2 * HID, 2 * HID, 1, stream);
        cvt(U2 + (size_t)i * HID * 2 * HID, wb, HID * 2 * HID, stream);
        run_gemm(xlb, wb, u2b + (size_t)i * HID, xloc, nullptr, N_NODES, HID, 2 * HID, 0, stream);

        // x = LN(x + xu); also emit bf16(x) for next layer / head
        ln_residual_kernel<<<N_NODES, HID, 0, stream>>>(x, xloc, nullptr,
                                                        g3 + (size_t)i * HID, be3 + (size_t)i * HID,
                                                        x, xb);
    }

    // head: out = gelu(x @ O1^T + o1b) @ O2^T + o2b
    cvt(O1, wb, HID * HID, stream);
    run_gemm(xb, wb, o1b, xloc, nullptr, N_NODES, HID, HID, 1, stream);
    final_logits_kernel<<<(N_NODES + 255) / 256, 256, 0, stream>>>(xloc, O2, o2b, (float*)d_out);
}
